// MultiHeadAttention_88519275970813
// MI455X (gfx1250) — matrix operand order repaired and checked by reading, not compiled, not run on hardware
//
#include <hip/hip_runtime.h>
#include <hip/hip_bf16.h>

// ---------------------------------------------------------------------------
// MultiHeadAttention for MI455X (gfx1250): bf16 WMMA everywhere,
// software-pipelined fragment loads, 32x64 wave tiles in the dense GEMMs.
// B=2, T=2048, C=1024, H=16, D=64.
// ---------------------------------------------------------------------------

typedef __bf16 bf16_t;
typedef __attribute__((ext_vector_type(16))) __bf16 v16bf;
typedef __attribute__((ext_vector_type(8)))  __bf16 v8bf;
typedef __attribute__((ext_vector_type(8)))  float  v8f;

#define BB   2
#define TT   2048
#define CC   1024
#define HH   16
#define DD   64
#define C3   3072
#define MTOT (BB*TT)          // 4096 rows of x

// ---- WMMA helpers ----------------------------------------------------------

__device__ __forceinline__ v8f wmma_bf16(v16bf a, v16bf b, v8f c) {
  // D = A(16x32) * B(32x16) + C, f32 accumulate
  return __builtin_amdgcn_wmma_f32_16x16x32_bf16(
      /*neg_a=*/false, a, /*neg_b=*/false, b,
      /*c_mod=*/(short)0, c, /*reuse_a=*/false, /*reuse_b=*/false);
}

__device__ __forceinline__ v16bf frag_cat(const bf16_t* p0, const bf16_t* p1) {
  v8bf lo = *(const v8bf*)p0;
  v8bf hi = *(const v8bf*)p1;
  return __builtin_shufflevector(lo, hi, 0,1,2,3,4,5,6,7,8,9,10,11,12,13,14,15);
}

// A fragment (16x32, 16-bit): lane m = lane&15; k = kb + 16*half + e (contig 16)
__device__ __forceinline__ v16bf load_a_frag(const bf16_t* row, int kb, int half) {
  return frag_cat(row + kb + 16*half, row + kb + 16*half + 8);
}
// B fragment (32x16, 16-bit): lane n = lane&15; k = kb + 16*half + e (contig 16)
__device__ __forceinline__ v16bf load_b_frag(const bf16_t* row, int kb, int half) {
  return frag_cat(row + kb + 16*half, row + kb + 16*half + 8);
}

// Software-pipelined 32x64 tile GEMM core over K=CC.
// arow0/arow1: A rows (lane&15 within row-groups m0 / m0+16), stride CC.
// brow: Wt row for this lane's column n0+l16, stride CC per 16-column block.
__device__ __forceinline__ void gemm_core_32x64(const bf16_t* arow0,
                                                const bf16_t* arow1,
                                                const bf16_t* brow,
                                                int half, v8f acc[8]) {
  v16bf a0 = load_a_frag(arow0, 0, half);
  v16bf a1 = load_a_frag(arow1, 0, half);
  v16bf b0 = load_b_frag(brow + (size_t)0  * CC, 0, half);
  v16bf b1 = load_b_frag(brow + (size_t)16 * CC, 0, half);
  v16bf b2 = load_b_frag(brow + (size_t)32 * CC, 0, half);
  v16bf b3 = load_b_frag(brow + (size_t)48 * CC, 0, half);
  for (int k = 0; k < CC; k += 32) {
    const int kn = (k + 32 < CC) ? k + 32 : 0;   // harmless reload on last iter
    v16bf na0 = load_a_frag(arow0, kn, half);
    v16bf na1 = load_a_frag(arow1, kn, half);
    v16bf nb0 = load_b_frag(brow + (size_t)0  * CC, kn, half);
    v16bf nb1 = load_b_frag(brow + (size_t)16 * CC, kn, half);
    v16bf nb2 = load_b_frag(brow + (size_t)32 * CC, kn, half);
    v16bf nb3 = load_b_frag(brow + (size_t)48 * CC, kn, half);
    acc[0] = wmma_bf16(a0, b0, acc[0]);
    acc[1] = wmma_bf16(a0, b1, acc[1]);
    acc[2] = wmma_bf16(a0, b2, acc[2]);
    acc[3] = wmma_bf16(a0, b3, acc[3]);
    acc[4] = wmma_bf16(a1, b0, acc[4]);
    acc[5] = wmma_bf16(a1, b1, acc[5]);
    acc[6] = wmma_bf16(a1, b2, acc[6]);
    acc[7] = wmma_bf16(a1, b3, acc[7]);
    a0 = na0; a1 = na1; b0 = nb0; b1 = nb1; b2 = nb2; b3 = nb3;
  }
}

// ---- Prep kernels ----------------------------------------------------------

__global__ void cvt_f32_to_bf16(const float* __restrict__ src,
                                bf16_t* __restrict__ dst, int n) {
  int i = blockIdx.x * blockDim.x + threadIdx.x;
  if (i < n) dst[i] = (bf16_t)src[i];
}

// Wt[n][k] = W[k][n]  (K x N row-major in, N x K row-major out), + f32->bf16
__global__ void transpose_to_bf16(const float* __restrict__ W,
                                  bf16_t* __restrict__ Wt, int K, int N) {
  int idx = blockIdx.x * blockDim.x + threadIdx.x;
  if (idx < K * N) {
    int k = idx / N;
    int n = idx - k * N;
    Wt[(size_t)n * K + k] = (bf16_t)W[idx];
  }
}

// ---- Kernel 2: fused QKV GEMM + head-major scatter -------------------------
// xb [4096, 1024] bf16, WtA [3072, 1024] bf16 (pre-transposed).
// Each wave: 32(M) x 64(N). Block = 4 waves along N (256 cols).
__global__ __launch_bounds__(128)
void qkv_gemm(const bf16_t* __restrict__ xb, const bf16_t* __restrict__ WtA,
              const float* __restrict__ b_attn,
              bf16_t* __restrict__ Qb, bf16_t* __restrict__ Kb,
              bf16_t* __restrict__ Vt) {
  const int lane = threadIdx.x & 31;
  const int wave = threadIdx.x >> 5;
  const int half = lane >> 4, l16 = lane & 15;
  const int m0 = blockIdx.y * 32;
  const int n0 = (blockIdx.x * 4 + wave) * 64;

  v8f acc[8] = {v8f{}, v8f{}, v8f{}, v8f{}, v8f{}, v8f{}, v8f{}, v8f{}};
  gemm_core_32x64(xb + (size_t)(m0 + l16) * CC,
                  xb + (size_t)(m0 + 16 + l16) * CC,
                  WtA + (size_t)(n0 + l16) * CC, half, acc);

#pragma unroll
  for (int i = 0; i < 4; ++i) {
    const int col = n0 + 16 * i + l16;          // 0..3071
    const float bias = b_attn[col];
    const int seg = col >> 10;                  // 0=q 1=k 2=v
    const int cj = col & (CC - 1);
    const int hh = cj >> 6, d = cj & 63;
#pragma unroll
    for (int g = 0; g < 2; ++g) {
#pragma unroll
      for (int r = 0; r < 8; ++r) {
        const int m = m0 + 16 * g + r + 8 * half;   // global token row
        const int b_ = m >> 11, t = m & (TT - 1);
        const size_t bh = (size_t)b_ * HH + hh;
        const bf16_t val = (bf16_t)(acc[4 * g + i][r] + bias);
        if (seg == 0)      Qb[(bh * TT + t) * DD + d] = val;
        else if (seg == 1) Kb[(bh * TT + t) * DD + d] = val;
        else               Vt[(bh * DD + d) * TT + t] = val;  // transposed V
      }
    }
  }
}

// ---- Kernel 3: causal attention (one wave = one 16-query tile) -------------
// LDS: per-wave 16 x 2048 bf16 score/prob rows (64 KB per wave).
__global__ __launch_bounds__(64)
void attn_kernel(const bf16_t* __restrict__ Qb, const bf16_t* __restrict__ Kb,
                 const bf16_t* __restrict__ Vt, bf16_t* __restrict__ att) {
  extern __shared__ char smem_raw[];
  __shared__ float redmax[2][16];
  __shared__ float redsum[2][16];

  const int lane = threadIdx.x & 31;
  const int wave = threadIdx.x >> 5;
  const int half = lane >> 4, l16 = lane & 15;
  bf16_t* S = (bf16_t*)smem_raw + (size_t)wave * 16 * TT;

  const int bh = blockIdx.y;                 // b*16 + h
  const int qt = blockIdx.x * 2 + wave;      // query tile 0..127
  const int q0 = qt * 16;

  const bf16_t* Qh = Qb + (size_t)bh * TT * DD;
  const bf16_t* Kh = Kb + (size_t)bh * TT * DD;
  const bf16_t* Vh = Vt + (size_t)bh * DD * TT;

  const float NEG = -3.0e38f;

  // ---- Pass 1: S = Q K^T (causal mask), track row max; pipelined K loads ----
  float rmax[8];
#pragma unroll
  for (int r = 0; r < 8; ++r) rmax[r] = NEG;

  const bf16_t* qrow = Qh + (size_t)(q0 + l16) * DD;
  const v16bf a0 = load_a_frag(qrow, 0, half);
  const v16bf a1 = load_a_frag(qrow, 32, half);

  const int nk = q0 + 16;                    // valid keys [0, nk)
  {
    const bf16_t* krow = Kh + (size_t)l16 * DD;
    v16bf b0 = load_b_frag(krow, 0, half);
    v16bf b1 = load_b_frag(krow, 32, half);
    for (int kt = 0; kt < nk; kt += 16) {
      const int ktn = (kt + 16 < nk) ? kt + 16 : kt;
      const bf16_t* nkrow = Kh + (size_t)(ktn + l16) * DD;
      v16bf nb0 = load_b_frag(nkrow, 0, half);
      v16bf nb1 = load_b_frag(nkrow, 32, half);

      v8f s = v8f{};
      s = wmma_bf16(a0, b0, s);
      s = wmma_bf16(a1, b1, s);

      const int col = kt + l16;
#pragma unroll
      for (int r = 0; r < 8; ++r) {
        const int row = q0 + r + 8 * half;
        float v = s[r];
        if (col > row) v = NEG;              // causal mask
        s[r] = v;
        float mx = v;
#pragma unroll
        for (int off = 1; off < 16; off <<= 1)
          mx = fmaxf(mx, __shfl_xor(mx, off, 32));
        rmax[r] = fmaxf(rmax[r], mx);
      }
#pragma unroll
      for (int r = 0; r < 8; ++r)
        S[(size_t)(r + 8 * half) * TT + col] = (bf16_t)s[r];

      b0 = nb0; b1 = nb1;
    }
  }

  // publish per-row max: lanes 0 / 16 cover rows 0-7 / 8-15
  if (l16 == 0) {
#pragma unroll
    for (int r = 0; r < 8; ++r) redmax[wave][8 * half + r] = rmax[r];
  }
  asm volatile("s_wait_dscnt 0" ::: "memory");

  // ---- Pass 2: softmax numerator into LDS (bf16), row sums ----
  const int nk32 = (nk + 31) & ~31;
  const float k2e = 0.03125f * 1.44269504088896340736f;  // (1/sqrt(C))*log2(e)
  const float mrow = redmax[wave][l16];
  bf16_t* srow = S + (size_t)l16 * TT;
  const int c0 = half * (nk32 >> 1);
  const int c1 = c0 + (nk32 >> 1);
  float lsum = 0.f;
  for (int c = c0; c < c1; ++c) {
    float p = 0.f;
    if (c < nk) {
      const float sv = (float)srow[c];
      p = exp2f((sv - mrow) * k2e);
      if (sv <= -1.0e38f) p = 0.f;
    }
    lsum += p;
    srow[c] = (bf16_t)p;
  }
  lsum += __shfl_xor(lsum, 16, 32);          // combine the two column halves
  if (half == 0) redsum[wave][l16] = lsum;
  asm volatile("s_wait_dscnt 0" ::: "memory");

  // ---- Pass 3: O = P V (P from LDS, V^T from global); pipelined loads ----
  v8f o[4] = {v8f{}, v8f{}, v8f{}, v8f{}};
  const bf16_t* prow = S + (size_t)l16 * TT;
  const bf16_t* vrow0 = Vh + (size_t)(0  + l16) * TT;
  const bf16_t* vrow1 = Vh + (size_t)(16 + l16) * TT;
  const bf16_t* vrow2 = Vh + (size_t)(32 + l16) * TT;
  const bf16_t* vrow3 = Vh + (size_t)(48 + l16) * TT;
  {
    v16bf pa = load_a_frag(prow, 0, half);
    v16bf vb0 = load_b_frag(vrow0, 0, half);
    v16bf vb1 = load_b_frag(vrow1, 0, half);
    v16bf vb2 = load_b_frag(vrow2, 0, half);
    v16bf vb3 = load_b_frag(vrow3, 0, half);
    for (int ks = 0; ks < nk32; ks += 32) {
      const int ksn = (ks + 32 < nk32) ? ks + 32 : 0;
      v16bf npa  = load_a_frag(prow, ksn, half);
      v16bf nvb0 = load_b_frag(vrow0, ksn, half);
      v16bf nvb1 = load_b_frag(vrow1, ksn, half);
      v16bf nvb2 = load_b_frag(vrow2, ksn, half);
      v16bf nvb3 = load_b_frag(vrow3, ksn, half);
      o[0] = wmma_bf16(pa, vb0, o[0]);
      o[1] = wmma_bf16(pa, vb1, o[1]);
      o[2] = wmma_bf16(pa, vb2, o[2]);
      o[3] = wmma_bf16(pa, vb3, o[3]);
      pa = npa; vb0 = nvb0; vb1 = nvb1; vb2 = nvb2; vb3 = nvb3;
    }
  }

  float inv[8];
#pragma unroll
  for (int r = 0; r < 8; ++r) inv[r] = 1.0f / redsum[wave][r + 8 * half];

  const int b_ = bh >> 4, hh = bh & 15;
#pragma unroll
  for (int i = 0; i < 4; ++i) {
    const int d = i * 16 + l16;
#pragma unroll
    for (int r = 0; r < 8; ++r) {
      const int t = q0 + r + 8 * half;
      att[((size_t)(b_ * TT + t)) * CC + hh * DD + d] = (bf16_t)(o[i][r] * inv[r]);
    }
  }
}

// ---- Kernel 4: output projection (bf16 x bf16 -> f32) ----------------------
__global__ __launch_bounds__(128)
void proj_gemm(const bf16_t* __restrict__ att, const bf16_t* __restrict__ WtP,
               const float* __restrict__ b_proj, float* __restrict__ out) {
  const int lane = threadIdx.x & 31;
  const int wave = threadIdx.x >> 5;
  const int half = lane >> 4, l16 = lane & 15;
  const int m0 = blockIdx.y * 32;
  const int n0 = (blockIdx.x * 4 + wave) * 64;

  v8f acc[8] = {v8f{}, v8f{}, v8f{}, v8f{}, v8f{}, v8f{}, v8f{}, v8f{}};
  gemm_core_32x64(att + (size_t)(m0 + l16) * CC,
                  att + (size_t)(m0 + 16 + l16) * CC,
                  WtP + (size_t)(n0 + l16) * CC, half, acc);

#pragma unroll
  for (int i = 0; i < 4; ++i) {
    const int col = n0 + 16 * i + l16;
    const float bias = b_proj[col];
#pragma unroll
    for (int g = 0; g < 2; ++g) {
#pragma unroll
      for (int r = 0; r < 8; ++r) {
        const int m = m0 + 16 * g + r + 8 * half;
        out[(size_t)m * CC + col] = acc[4 * g + i][r] + bias;
      }
    }
  }
}

// ---------------------------------------------------------------------------

extern "C" void kernel_launch(void* const* d_in, const int* in_sizes, int n_in,
                              void* d_out, int out_size, void* d_ws, size_t ws_size,
                              hipStream_t stream) {
  const float* x      = (const float*)d_in[0];  // [B,T,C]
  const float* W_attn = (const float*)d_in[1];  // [C,3C]
  const float* b_attn = (const float*)d_in[2];  // [3C]
  const float* W_proj = (const float*)d_in[3];  // [C,C]
  const float* b_proj = (const float*)d_in[4];  // [C]
  float* out = (float*)d_out;                   // [B,T,C]

  // Workspace partition (bf16 elements), total ~50.3 MB
  bf16_t* p = (bf16_t*)d_ws;
  bf16_t* xb  = p; p += (size_t)MTOT * CC;      // 4096x1024
  bf16_t* WtA = p; p += (size_t)C3 * CC;        // 3072x1024 (transposed)
  bf16_t* WtP = p; p += (size_t)CC * CC;        // 1024x1024 (transposed)
  bf16_t* Qb  = p; p += (size_t)BB * HH * TT * DD;  // [B,H,T,D]
  bf16_t* Kb  = p; p += (size_t)BB * HH * TT * DD;  // [B,H,T,D]
  bf16_t* Vt  = p; p += (size_t)BB * HH * DD * TT;  // [B,H,D,T] (transposed)
  bf16_t* att = p; p += (size_t)MTOT * CC;      // [B,T,C] attention output

  // 1) precision / layout prep
  {
    int n = MTOT * CC;
    cvt_f32_to_bf16<<<(n + 255) / 256, 256, 0, stream>>>(x, xb, n);
  }
  {
    int n = CC * C3;
    transpose_to_bf16<<<(n + 255) / 256, 256, 0, stream>>>(W_attn, WtA, CC, C3);
  }
  {
    int n = CC * CC;
    transpose_to_bf16<<<(n + 255) / 256, 256, 0, stream>>>(W_proj, WtP, CC, CC);
  }

  // 2) QKV GEMM: M=4096 (32/block-row), N=3072 (256/block)
  qkv_gemm<<<dim3(C3 / 256, MTOT / 32), 128, 0, stream>>>(
      xb, WtA, b_attn, Qb, Kb, Vt);

  // 3) attention: 2 waves/block, 128 KB dynamic LDS (16x2048 bf16 per wave)
  const size_t attn_lds = (size_t)2 * 16 * TT * sizeof(bf16_t);
  (void)hipFuncSetAttribute((const void*)attn_kernel,
                            hipFuncAttributeMaxDynamicSharedMemorySize,
                            (int)attn_lds);
  attn_kernel<<<dim3(TT / 16 / 2, BB * HH), 64, attn_lds, stream>>>(
      Qb, Kb, Vt, att);

  // 4) output projection: M=4096, N=1024
  proj_gemm<<<dim3(CC / 256, MTOT / 32), 128, 0, stream>>>(
      att, WtP, b_proj, out);
}
